// TransactionGNN_69887707840602
// MI455X (gfx1250) — matrix operand order repaired and checked by reading, not compiled, not run on hardware
//
#include <hip/hip_runtime.h>
#include <hip/hip_bf16.h>

// ---------------------------------------------------------------------------
// CDNA5 (gfx1250) GAT pipeline, atomic-free aggregation via dst-CSR:
//   x = emb[y] -> 2 x ( WMMA f16 GEMM -> attn logits -> per-dst wave:
//                       online softmax + weighted gather + ELU + L2 norm )
//   -> MLP head on last row.
// Matrix ops: v_wmma_f32_16x16x32_f16 (wave32 WMMA, f32 accumulate),
// B fragments persisted in VGPRs across 8 M-tiles per wave.
// Node-row buffers padded to a multiple of 16 rows -> GEMM has no guards.
// ---------------------------------------------------------------------------

typedef __attribute__((ext_vector_type(16))) _Float16 v16h;
typedef __attribute__((ext_vector_type(8)))  _Float16 v8h;
typedef __attribute__((ext_vector_type(8)))  float    v8f;

#define HD 256      // HEADS*OUT
#define NHEADS 4
#define CHANNELS 64
#define NEG_SLOPE 0.2f

__device__ __forceinline__ float lrelu(float e) {
  return e > 0.f ? e : NEG_SLOPE * e;
}
// online softmax: accumulate one sample
__device__ __forceinline__ void onl_acc(float& m, float& s, float e) {
  float nm = fmaxf(m, e);
  s = s * __expf(m - nm) + __expf(e - nm);
  m = nm;
}
// online softmax: combine two partial states
__device__ __forceinline__ void onl_comb(float& m, float& s, float mo, float so) {
  float nm = fmaxf(m, mo);
  s = s * __expf(m - nm) + so * __expf(mo - nm);
  m = nm;
}

// ---------------------------------------------------------------------------
// 0) Gather rows of emb into f16 X0 [N,64]
// ---------------------------------------------------------------------------
__global__ void k_gather(const int* __restrict__ y, const float* __restrict__ emb,
                         _Float16* __restrict__ xh, int N) {
  int tid = blockIdx.x * blockDim.x + threadIdx.x;
  if (tid >= N * CHANNELS) return;
  int n = tid >> 6, k = tid & 63;
  xh[tid] = (_Float16)emb[(size_t)y[n] * CHANNELS + k];
}

// ---------------------------------------------------------------------------
// 1) CSR build: zero, degree histogram, single-block scan, scatter
// ---------------------------------------------------------------------------
__global__ void k_zero2(int* __restrict__ a, int* __restrict__ b, int N) {
  int i = blockIdx.x * blockDim.x + threadIdx.x;
  if (i < N) { a[i] = 0; b[i] = 0; }
}

__global__ void k_deg(const int* __restrict__ ei, int* __restrict__ deg,
                      int E, int Etot) {
  int i = blockIdx.x * blockDim.x + threadIdx.x;
  if (i >= Etot) return;
  int d = (i < E) ? ei[E + i] : i - E;
  atomicAdd(&deg[d], 1);
}

__global__ void k_scan(const int* __restrict__ deg, int* __restrict__ rowstart,
                       int N) {
  __shared__ int sh[1024];
  __shared__ int running;
  int t = threadIdx.x;
  if (t == 0) running = 0;
  __syncthreads();
  for (int base = 0; base < N; base += 1024) {
    int v = (base + t < N) ? deg[base + t] : 0;
    sh[t] = v;
    __syncthreads();
    for (int off = 1; off < 1024; off <<= 1) {
      int x = (t >= off) ? sh[t - off] : 0;
      __syncthreads();
      sh[t] += x;
      __syncthreads();
    }
    int incl  = sh[t];
    int total = sh[1023];
    int roff  = running;
    __syncthreads();
    if (base + t < N) rowstart[base + t] = roff + incl - v;
    if (t == 0) running += total;
    __syncthreads();
  }
  if (t == 0) rowstart[N] = running;
}

__global__ void k_scatter(const int* __restrict__ ei,
                          const int* __restrict__ rowstart,
                          int* __restrict__ cursor, int* __restrict__ srcSorted,
                          int E, int Etot) {
  int i = blockIdx.x * blockDim.x + threadIdx.x;
  if (i >= Etot) return;
  int s, d;
  if (i < E) { s = ei[i]; d = ei[E + i]; } else { s = d = i - E; }
  int pos = atomicAdd(&cursor[d], 1);
  srcSorted[rowstart[d] + pos] = s;
}

// ---------------------------------------------------------------------------
// 2) Transpose W [K,256] (f32) -> Wt [256,K] (f16)
// ---------------------------------------------------------------------------
__global__ void k_transposeW(const float* __restrict__ W, _Float16* __restrict__ Wt,
                             int K) {
  int tid = blockIdx.x * blockDim.x + threadIdx.x;
  if (tid >= HD * K) return;
  int n = tid / K, k = tid % K;
  Wt[(size_t)n * K + k] = (_Float16)W[(size_t)k * HD + n];
}

// ---------------------------------------------------------------------------
// 3) WMMA GEMM: H[Mtiles*16,256] = Xh[Mtiles*16,K] * Wt[256,K]^T
//    Buffers padded to full 16-row tiles: no bounds guards anywhere.
//    B fragments persisted in registers; each wave sweeps 8 M-tiles for one
//    16-wide N-tile. blockDim = 128 (4 waves), grid = ((Mtiles+7)/8, 4).
// ---------------------------------------------------------------------------
template <int K>
__global__ void k_gemm_f16(const _Float16* __restrict__ Xh,
                           const _Float16* __restrict__ Wt,
                           float* __restrict__ H, int Mtiles) {
  constexpr int NCH = K / 32;
  const int lane = threadIdx.x & 31;
  const int g    = lane >> 4;
  const int r    = lane & 15;
  const int n0   = (blockIdx.y * 4 + (threadIdx.x >> 5)) * 16;
  const int mt0  = blockIdx.x * 8;

  // persist all B fragments for this 16-col tile in VGPRs
  v16h bfrag[NCH];
  const _Float16* brow = Wt + (size_t)(n0 + r) * K + 16 * g;
#pragma unroll
  for (int ch = 0; ch < NCH; ++ch)
    bfrag[ch] = *reinterpret_cast<const v16h*>(brow + ch * 32);

  const int col = n0 + r;
  for (int t = 0; t < 8; ++t) {
    int mtile = mt0 + t;
    if (mtile >= Mtiles) break;
    const _Float16* arow = Xh + (size_t)(mtile * 16 + r) * K;
    __builtin_prefetch(arow + K, 0, 0);          // next row, global_prefetch_b8
    v8f c = {};
#pragma unroll
    for (int ch = 0; ch < NCH; ++ch) {
      v8h a0 = *reinterpret_cast<const v8h*>(arow + ch * 32 + 16 * g);
      v8h a1 = *reinterpret_cast<const v8h*>(arow + ch * 32 + 16 * g + 8);
      v16h a;
#pragma unroll
      for (int i = 0; i < 8; ++i) { a[i] = a0[i]; a[i + 8] = a1[i]; }
      c = __builtin_amdgcn_wmma_f32_16x16x32_f16(
          false, a, false, bfrag[ch], (short)0, c, false, false);
    }
    // D layout: VGPR i -> row mtile*16 + i + 8*g, col n0 + r
    float* hp = H + (size_t)(mtile * 16 + 8 * g) * HD + col;
#pragma unroll
    for (int i = 0; i < 8; ++i) hp[(size_t)i * HD] = c[i];   // one store clause
  }
}

// ---------------------------------------------------------------------------
// 4) Attention logits: al_s[n,h] = sum_c H[n,h*64+c]*a_src[h*64+c]
// ---------------------------------------------------------------------------
__global__ void k_attn_logits(const float* __restrict__ H,
                              const float* __restrict__ a_src,
                              const float* __restrict__ a_dst,
                              float* __restrict__ als, float* __restrict__ ald,
                              int N) {
  int n = (blockIdx.x * blockDim.x + threadIdx.x) >> 5;
  if (n >= N) return;
  int lane = threadIdx.x & 31;
  float accs[NHEADS] = {0.f, 0.f, 0.f, 0.f};
  float accd[NHEADS] = {0.f, 0.f, 0.f, 0.f};
#pragma unroll
  for (int e = 0; e < 8; ++e) {
    int j = lane + 32 * e;
    float v = H[(size_t)n * HD + j];
    accs[e >> 1] += v * a_src[j];
    accd[e >> 1] += v * a_dst[j];
  }
#pragma unroll
  for (int h = 0; h < NHEADS; ++h) {
#pragma unroll
    for (int off = 16; off > 0; off >>= 1) {
      accs[h] += __shfl_xor(accs[h], off, 32);
      accd[h] += __shfl_xor(accd[h], off, 32);
    }
  }
  if (lane == 0) {
    reinterpret_cast<float4*>(als)[n] = make_float4(accs[0], accs[1], accs[2], accs[3]);
    reinterpret_cast<float4*>(ald)[n] = make_float4(accd[0], accd[1], accd[2], accd[3]);
  }
}

// ---------------------------------------------------------------------------
// 5) Fused per-dst aggregation: online softmax over incoming edges, weighted
//    gather of h[src], + bias, ELU, L2-normalize, emit f16 (+ ctx row).
//    One wave per destination node; zero atomics.
// ---------------------------------------------------------------------------
__global__ void k_aggregate(const int* __restrict__ rowstart,
                            const int* __restrict__ srcSorted,
                            const float* __restrict__ als,
                            const float* __restrict__ ald,
                            const float* __restrict__ H,
                            const float* __restrict__ bias,
                            _Float16* __restrict__ xh_next,
                            float* __restrict__ ctx, int N, int writeCtx) {
  int n = (blockIdx.x * blockDim.x + threadIdx.x) >> 5;
  if (n >= N) return;
  int lane  = threadIdx.x & 31;
  int start = rowstart[n];
  int end   = rowstart[n + 1];   // deg >= 1 (self loop)
  float4 ad = reinterpret_cast<const float4*>(ald)[n];

  // ---- pass 1: lane-parallel online softmax stats per head ----
  float m0 = -1e30f, m1 = -1e30f, m2 = -1e30f, m3 = -1e30f;
  float s0 = 0.f, s1 = 0.f, s2 = 0.f, s3 = 0.f;
  for (int idx = start + lane; idx < end; idx += 32) {
    float4 as = reinterpret_cast<const float4*>(als)[srcSorted[idx]];
    onl_acc(m0, s0, lrelu(as.x + ad.x));
    onl_acc(m1, s1, lrelu(as.y + ad.y));
    onl_acc(m2, s2, lrelu(as.z + ad.z));
    onl_acc(m3, s3, lrelu(as.w + ad.w));
  }
#pragma unroll
  for (int off = 16; off > 0; off >>= 1) {
    onl_comb(m0, s0, __shfl_xor(m0, off, 32), __shfl_xor(s0, off, 32));
    onl_comb(m1, s1, __shfl_xor(m1, off, 32), __shfl_xor(s1, off, 32));
    onl_comb(m2, s2, __shfl_xor(m2, off, 32), __shfl_xor(s2, off, 32));
    onl_comb(m3, s3, __shfl_xor(m3, off, 32), __shfl_xor(s3, off, 32));
  }
  float di0 = 1.f / (s0 + 1e-16f), di1 = 1.f / (s1 + 1e-16f);
  float di2 = 1.f / (s2 + 1e-16f), di3 = 1.f / (s3 + 1e-16f);

  // ---- pass 2: weighted gather, acc in registers ----
  float acc[8];
#pragma unroll
  for (int e = 0; e < 8; ++e) acc[e] = bias[lane + 32 * e];

  for (int base = start; base < end; base += 32) {
    int cnt = end - base; if (cnt > 32) cnt = 32;
    int idx = base + lane; if (idx >= end) idx = end - 1;
    int sl  = srcSorted[idx];
    float4 as = reinterpret_cast<const float4*>(als)[sl];
    float a0 = __expf(lrelu(as.x + ad.x) - m0) * di0;
    float a1 = __expf(lrelu(as.y + ad.y) - m1) * di1;
    float a2 = __expf(lrelu(as.z + ad.z) - m2) * di2;
    float a3 = __expf(lrelu(as.w + ad.w) - m3) * di3;
    for (int j = 0; j < cnt; ++j) {
      int   sj = __shfl(sl, j, 32);
      float aj[4] = {__shfl(a0, j, 32), __shfl(a1, j, 32),
                     __shfl(a2, j, 32), __shfl(a3, j, 32)};
      const float* hrow = H + (size_t)sj * HD;
#pragma unroll
      for (int e = 0; e < 8; ++e)
        acc[e] += hrow[lane + 32 * e] * aj[e >> 1];
    }
  }

  // ---- fused epilogue: ELU -> L2 normalize -> f16 ----
  float ss = 0.f;
#pragma unroll
  for (int e = 0; e < 8; ++e) {
    float v = acc[e];
    v = v > 0.f ? v : (__expf(v) - 1.f);
    acc[e] = v;
    ss += v * v;
  }
#pragma unroll
  for (int off = 16; off > 0; off >>= 1) ss += __shfl_xor(ss, off, 32);
  float inv = 1.f / fmaxf(__builtin_sqrtf(ss), 1e-12f);
#pragma unroll
  for (int e = 0; e < 8; ++e) {
    int j = lane + 32 * e;
    float v = acc[e] * inv;
    xh_next[(size_t)n * HD + j] = (_Float16)v;
    if (writeCtx && n == N - 1) ctx[j] = v;
  }
}

// ---------------------------------------------------------------------------
// 6) MLP head
// ---------------------------------------------------------------------------
__global__ void k_head1(const float* __restrict__ ctx,
                        const float* __restrict__ pw1,
                        const float* __restrict__ pb1,
                        float* __restrict__ h1) {
  int t = threadIdx.x;   // 0..63
  float acc = pb1[t];
  for (int k = 0; k < HD; ++k) acc += ctx[k] * pw1[(size_t)k * CHANNELS + t];
  h1[t] = fmaxf(acc, 0.f);
}

__global__ void k_head2(const float* __restrict__ h1,
                        const float* __restrict__ pw2,
                        const float* __restrict__ pb2,
                        float* __restrict__ outv, int V) {
  __shared__ float sh1[CHANNELS];
  if (threadIdx.x < CHANNELS) sh1[threadIdx.x] = h1[threadIdx.x];
  __syncthreads();
  int v = blockIdx.x * blockDim.x + threadIdx.x;
  if (v >= V) return;
  float acc = pb2[v];
#pragma unroll 8
  for (int dIdx = 0; dIdx < CHANNELS; ++dIdx)
    acc += sh1[dIdx] * pw2[(size_t)dIdx * V + v];
  outv[v] = acc;
}

// ---------------------------------------------------------------------------
// Host side
// ---------------------------------------------------------------------------
static inline size_t alignUp(size_t x) { return (x + 255) & ~(size_t)255; }

extern "C" void kernel_launch(void* const* d_in, const int* in_sizes, int n_in,
                              void* d_out, int out_size, void* d_ws, size_t ws_size,
                              hipStream_t stream) {
  const int*   y    = (const int*)d_in[0];
  const int*   ei   = (const int*)d_in[1];
  const float* emb  = (const float*)d_in[2];
  const float* Wm[2]    = {(const float*)d_in[3], (const float*)d_in[7]};
  const float* a_src[2] = {(const float*)d_in[4], (const float*)d_in[8]};
  const float* a_dst[2] = {(const float*)d_in[5], (const float*)d_in[9]};
  const float* bias[2]  = {(const float*)d_in[6], (const float*)d_in[10]};
  const float* pw1 = (const float*)d_in[11];
  const float* pb1 = (const float*)d_in[12];
  const float* pw2 = (const float*)d_in[13];
  const float* pb2 = (const float*)d_in[14];

  const int N      = in_sizes[0];
  const int E      = in_sizes[1] / 2;
  const int V      = in_sizes[14];
  const int Etot   = E + N;
  const int Mtiles = (N + 15) / 16;
  const int Mpad   = Mtiles * 16;    // padded row count -> guard-free GEMM

  // ---- carve workspace (row buffers padded to Mpad rows) ----
  char* ws = (char*)d_ws;
  size_t off = 0;
  auto carve = [&](size_t bytes) { char* p = ws + off; off += alignUp(bytes); return p; };
  _Float16* xh0   = (_Float16*)carve((size_t)Mpad * CHANNELS * sizeof(_Float16));
  _Float16* xh1   = (_Float16*)carve((size_t)Mpad * HD * sizeof(_Float16));
  _Float16* Wt    = (_Float16*)carve((size_t)HD * HD * sizeof(_Float16));
  float*    hbuf  = (float*)carve((size_t)Mpad * HD * sizeof(float));
  float*    als   = (float*)carve((size_t)N * NHEADS * sizeof(float));
  float*    ald   = (float*)carve((size_t)N * NHEADS * sizeof(float));
  int*      deg   = (int*)carve((size_t)N * sizeof(int));
  int*      rowst = (int*)carve((size_t)(N + 1) * sizeof(int));
  int*      curs  = (int*)carve((size_t)N * sizeof(int));
  int*      srcS  = (int*)carve((size_t)Etot * sizeof(int));
  float*    ctx   = (float*)carve(HD * sizeof(float));
  float*    h1    = (float*)carve(CHANNELS * sizeof(float));
  (void)ws_size; (void)n_in; (void)out_size;

  // ---- gather embeddings (f16) + build dst-CSR (shared by both layers) ----
  k_gather<<<(N * CHANNELS + 255) / 256, 256, 0, stream>>>(y, emb, xh0, N);
  k_zero2<<<(N + 255) / 256, 256, 0, stream>>>(deg, curs, N);
  k_deg<<<(Etot + 255) / 256, 256, 0, stream>>>(ei, deg, E, Etot);
  k_scan<<<1, 1024, 0, stream>>>(deg, rowst, N);
  k_scatter<<<(Etot + 255) / 256, 256, 0, stream>>>(ei, rowst, curs, srcS, E, Etot);

  const _Float16* xin = xh0;
  for (int l = 0; l < 2; ++l) {
    const int K = (l == 0) ? CHANNELS : HD;
    k_transposeW<<<(HD * K + 255) / 256, 256, 0, stream>>>(Wm[l], Wt, K);
    dim3 gg((Mtiles + 7) / 8, 4);
    if (l == 0) k_gemm_f16<CHANNELS><<<gg, 128, 0, stream>>>(xin, Wt, hbuf, Mtiles);
    else        k_gemm_f16<HD>      <<<gg, 128, 0, stream>>>(xin, Wt, hbuf, Mtiles);
    k_attn_logits<<<(N + 7) / 8, 256, 0, stream>>>(hbuf, a_src[l], a_dst[l],
                                                   als, ald, N);
    k_aggregate<<<(N + 7) / 8, 256, 0, stream>>>(rowst, srcS, als, ald, hbuf,
                                                 bias[l], xh1, ctx, N, l == 1);
    xin = xh1;
  }

  k_head1<<<1, CHANNELS, 0, stream>>>(ctx, pw1, pb1, h1);
  k_head2<<<(V + 255) / 256, 256, 0, stream>>>(h1, pw2, pb2, (float*)d_out, V);
}
